// MultiheadAttention_13391708029496
// MI455X (gfx1250) — matrix operand order repaired and checked by reading, not compiled, not run on hardware
//
#include <hip/hip_runtime.h>
#include <hip/hip_bf16.h>

// ---------------------------------------------------------------------------
// MI455X (gfx1250) fused multi-head attention, wave32 + v_wmma_f32_16x16x32_f16
//   B=4 S=2048 QF=512 KF=VF=64 H=8
//   ~48 GFLOP vs ~52MB HBM traffic -> compute bound -> all GEMMs on WMMA.
//   - f32->f16 conversion hoisted into one-time convert/swizzle kernels
//   - GEMM loops double-buffered so WMMAs don't sit behind s_wait_loadcnt 0
//   - attention prefetches V across the softmax + P-transpose latency
// ---------------------------------------------------------------------------

typedef __attribute__((ext_vector_type(16))) _Float16 v16h;
typedef __attribute__((ext_vector_type(8)))  _Float16 v8h;
typedef __attribute__((ext_vector_type(4)))  _Float16 v4h;
typedef __attribute__((ext_vector_type(8)))  float    v8f;
typedef __attribute__((ext_vector_type(4)))  float    v4f;

constexpr int Bb = 4;
constexpr int Hh = 8;
constexpr int Ss = 2048;
constexpr int QF = 512;
constexpr int Dd = 64;     // KF == VF == 64

// ---- WMMA operand loaders (contiguous k order, used for both A and B) -------------------
// A-matrix 16x32 f16: lane holds row m = lane%16; element e holds
//   K = (lane/16)*16 + e
// => one contiguous 16-element run at (half*16).

__device__ inline v16h load_a_f16(const _Float16* __restrict__ base, int ld) {
  const int lane = threadIdx.x & 31;
  const int m = lane & 15, hf = lane >> 4;
  const _Float16* r = base + (size_t)m * ld + hf * 16;
  v8h lo = *(const v8h*)(r);
  v8h hi = *(const v8h*)(r + 8);
  v16h a;
#pragma unroll
  for (int i = 0; i < 8; ++i) { a[i] = lo[i]; a[8 + i] = hi[i]; }
  return a;
}

#define WMMA_F16(A, Bm, C) \
  __builtin_amdgcn_wmma_f32_16x16x32_f16(false, (A), false, (Bm), (short)0, (C), false, false)

// ---------------------------------------------------------------------------
// Kernel 0a: f32 -> f16 elementwise convert (activations), 4 elems/thread.
// ---------------------------------------------------------------------------
__global__ __launch_bounds__(256) void cvt_f16_kernel(
    const float* __restrict__ in, _Float16* __restrict__ out)
{
  const size_t i = ((size_t)blockIdx.x * blockDim.x + threadIdx.x) * 4;
  v4f x = *(const v4f*)(in + i);
  v4h y;
#pragma unroll
  for (int j = 0; j < 4; ++j) y[j] = (_Float16)x[j];
  *(v4h*)(out + i) = y;
}

// ---------------------------------------------------------------------------
// Kernel 0b: weight swizzle into WMMA B-operand order.
//   W: [nmat][512][64] f32  ->  Wsw: [nmat][k0/32][half][n][e] f16
//   Wsw linear index == gid:  e=b0..3  n=b4..9  hf=b10  kb=b11..14  mat=b15..
//   B tile load then = one aligned 32B v16h per lane, coalesced across lanes.
// ---------------------------------------------------------------------------
__global__ __launch_bounds__(256) void wswz_kernel(
    const float* __restrict__ W, _Float16* __restrict__ out)
{
  const size_t gid = (size_t)blockIdx.x * blockDim.x + threadIdx.x;
  const int e   = gid & 15;
  const int n   = (gid >> 4) & 63;
  const int hf  = (gid >> 10) & 1;
  const int kb  = (gid >> 11) & 15;
  const size_t mat = gid >> 15;
  const int k = kb * 32 + hf * 16 + e;
  out[gid] = (_Float16)W[(mat * QF + k) * Dd + n];
}

// ---------------------------------------------------------------------------
// Kernel 1: per-head input projection  out[b,h,s,d] (or transposed [b,h,d,s])
//   GEMM [8192 x 512] x [512 x 512(=H*64)], one wave = 16 rows x 64 cols.
//   Double-buffered: next k-step's A/B tiles are in flight during WMMAs.
// ---------------------------------------------------------------------------
__global__ __launch_bounds__(256) void proj_kernel(
    const _Float16* __restrict__ X,    // [B*S, QF] f16
    const _Float16* __restrict__ Wsw,  // swizzled f16 weights
    _Float16* __restrict__ out,        // f16 workspace
    float scale, int transpose)
{
  const int wid  = (blockIdx.x * blockDim.x + threadIdx.x) >> 5;
  const int lane = threadIdx.x & 31;
  const int col  = lane & 15, hf = lane >> 4;
  const int h    = wid & 7;          // head == 64-wide N tile
  const int r0   = (wid >> 3) * 16;  // 512 row tiles

  const _Float16* Xr = X + (size_t)r0 * QF;
  const _Float16* Wh = Wsw + ((size_t)h * 16 * 2 + hf) * 1024;  // + hf*64*16

  v8f acc[4] = {};
  v16h a = load_a_f16(Xr, QF);
  v16h b[4];
#pragma unroll
  for (int t = 0; t < 4; ++t)
    b[t] = *(const v16h*)(Wh + (t * 16 + col) * 16);

#pragma unroll
  for (int k0 = 0; k0 < QF; k0 += 32) {
    const int kn = (k0 + 32 < QF) ? (k0 + 32) : k0;  // last iter: dummy reload
    // prefetch next tiles first so WMMAs below overlap the loads
    v16h an = load_a_f16(Xr + kn, QF);
    const _Float16* wn = Wh + (size_t)(kn >> 5) * 2048;  // (kn/32)*2*64*16
    v16h bn[4];
#pragma unroll
    for (int t = 0; t < 4; ++t)
      bn[t] = *(const v16h*)(wn + (t * 16 + col) * 16);
#pragma unroll
    for (int t = 0; t < 4; ++t) acc[t] = WMMA_F16(a, b[t], acc[t]);
    a = an;
#pragma unroll
    for (int t = 0; t < 4; ++t) b[t] = bn[t];
  }

  // C tile: row = r0 + j + half*8, col = t*16 + lane%16
#pragma unroll
  for (int t = 0; t < 4; ++t) {
#pragma unroll
    for (int j = 0; j < 8; ++j) {
      int r = r0 + j + hf * 8;
      int bidx = r >> 11, s = r & (Ss - 1);
      int d = t * 16 + col;
      _Float16 val = (_Float16)(acc[t][j] * scale);
      if (transpose)
        out[(((size_t)bidx * Hh + h) * Dd + d) * Ss + s] = val;  // vT[b,h,d,s]
      else
        out[(((size_t)bidx * Hh + h) * Ss + s) * Dd + d] = val;  // q/k[b,h,s,d]
    }
  }
}

// ---------------------------------------------------------------------------
// Kernel 2: flash attention. Block = 8 waves = 128 query rows of one (b,h).
//   Wave: 16 q rows; KV tiles of 64; online softmax in f32.
//   V tiles are issued right after the K loads (in-order completion keeps the
//   score-WMMA waits off V) and land during softmax + P->LDS transpose.
// ---------------------------------------------------------------------------
__global__ __launch_bounds__(256) void attn_kernel(
    const _Float16* __restrict__ q,   // [b,h,s,64], pre-scaled by 1/8
    const _Float16* __restrict__ k,   // [b,h,s,64]
    const _Float16* __restrict__ vT,  // [b,h,64,s]
    _Float16* __restrict__ heads)     // [b,s,h*64+v]
{
  constexpr int PSTR = 72;  // 36-dword row stride -> conflict-free b128 LDS loads
  __shared__ alignas(16) _Float16 plds[8 * 16 * PSTR];

  const int wave = threadIdx.x >> 5;
  const int lane = threadIdx.x & 31;
  const int col  = lane & 15, hf = lane >> 4;
  const int qblk = blockIdx.x & 15;
  const int bh   = blockIdx.x >> 4;         // b*H + h
  const int b    = bh >> 3, h = bh & 7;
  const int qr0  = qblk * 128 + wave * 16;

  const _Float16* qbh = q  + (size_t)bh * Ss * Dd;
  const _Float16* kbh = k  + (size_t)bh * Ss * Dd;
  const _Float16* vbh = vT + (size_t)bh * Dd * Ss;
  _Float16* my = plds + wave * 16 * PSTR;   // wave-private P tile

  v16h aq[2];
#pragma unroll
  for (int dsl = 0; dsl < 2; ++dsl)
    aq[dsl] = load_a_f16(qbh + (size_t)qr0 * Dd + dsl * 32, Dd);

  v8f ov[4] = {};
  float mr[8], lr[8];
#pragma unroll
  for (int j = 0; j < 8; ++j) { mr[j] = -1e30f; lr[j] = 0.0f; }

  for (int kv0 = 0; kv0 < Ss; kv0 += 64) {
    // ---- scores S = Q K^T (pre-scaled): 4 key subtiles x 2 d-slices --------
    v8f sc[4];
#pragma unroll
    for (int t = 0; t < 4; ++t) {
      v8f c = {};
#pragma unroll
      for (int dsl = 0; dsl < 2; ++dsl) {
        // B(kk=d, n=key): contiguous row of K => direct aligned v16h load
        const _Float16* kr =
            kbh + (size_t)(kv0 + t * 16 + col) * Dd + dsl * 32 + hf * 16;
        v16h bm = *(const v16h*)kr;
        c = WMMA_F16(aq[dsl], bm, c);
      }
      sc[t] = c;
    }

    // ---- prefetch V tiles now; they complete during softmax ----------------
    v16h vbuf[8];
#pragma unroll
    for (int t = 0; t < 4; ++t)
#pragma unroll
      for (int ksl = 0; ksl < 2; ++ksl)
        vbuf[t * 2 + ksl] = *(const v16h*)(
            vbh + (size_t)(t * 16 + col) * Ss + kv0 + ksl * 32 + hf * 16);

    // ---- online softmax: row = j + half*8 lives in one 16-lane half --------
    float alpha[8];
#pragma unroll
    for (int j = 0; j < 8; ++j) {
      float m0 = fmaxf(fmaxf(sc[0][j], sc[1][j]), fmaxf(sc[2][j], sc[3][j]));
#pragma unroll
      for (int mk = 1; mk <= 8; mk <<= 1)
        m0 = fmaxf(m0, __shfl_xor(m0, mk, 32));
      float mn = fmaxf(mr[j], m0);
      alpha[j] = __expf(mr[j] - mn);
      mr[j] = mn;
    }
    float rs[8];
#pragma unroll
    for (int j = 0; j < 8; ++j) rs[j] = 0.0f;
#pragma unroll
    for (int t = 0; t < 4; ++t)
#pragma unroll
      for (int j = 0; j < 8; ++j) {
        float p = __expf(sc[t][j] - mr[j]);
        sc[t][j] = p;
        rs[j] += p;
      }
#pragma unroll
    for (int j = 0; j < 8; ++j) {
      float r = rs[j];
#pragma unroll
      for (int mk = 1; mk <= 8; mk <<= 1)
        r += __shfl_xor(r, mk, 32);
      lr[j] = lr[j] * alpha[j] + r;
    }
#pragma unroll
    for (int t = 0; t < 4; ++t)
#pragma unroll
      for (int j = 0; j < 8; ++j) ov[t][j] *= alpha[j];

    // ---- C-layout P -> LDS -> A-layout (wave-private, in-order DS ops) -----
#pragma unroll
    for (int t = 0; t < 4; ++t)
#pragma unroll
      for (int j = 0; j < 8; ++j)
        my[(j + hf * 8) * PSTR + t * 16 + col] = (_Float16)sc[t][j];

    v16h aP[2];
#pragma unroll
    for (int ksl = 0; ksl < 2; ++ksl)
      aP[ksl] = load_a_f16(my + ksl * 32, PSTR);

    // ---- O += P V : operands already resident ------------------------------
#pragma unroll
    for (int t = 0; t < 4; ++t) {
      v8f o = ov[t];
#pragma unroll
      for (int ksl = 0; ksl < 2; ++ksl)
        o = WMMA_F16(aP[ksl], vbuf[t * 2 + ksl], o);
      ov[t] = o;
    }
  }

#pragma unroll
  for (int j = 0; j < 8; ++j) lr[j] = 1.0f / lr[j];
#pragma unroll
  for (int t = 0; t < 4; ++t)
#pragma unroll
    for (int j = 0; j < 8; ++j) {
      int s = qr0 + j + hf * 8;
      heads[((size_t)b * Ss + s) * (Hh * Dd) + h * Dd + t * 16 + col] =
          (_Float16)(ov[t][j] * lr[j]);
    }
}

// ---------------------------------------------------------------------------
// Kernel 3: output projection.  sum_h heads[b,h,s,:] @ WO[h] ==
//   [8192 x 512(f16)] @ stackedWO[512 x 64]  -> out f32 [8192 x 64]
//   Same double-buffered structure as proj_kernel.
// ---------------------------------------------------------------------------
__global__ __launch_bounds__(256) void oproj_kernel(
    const _Float16* __restrict__ heads,  // [B*S, H*64]
    const _Float16* __restrict__ Wsw,    // swizzled f16 stacked WO [512x64]
    float* __restrict__ outp)            // [B*S, 64]
{
  const int wid  = (blockIdx.x * blockDim.x + threadIdx.x) >> 5;
  const int lane = threadIdx.x & 31;
  const int col  = lane & 15, hf = lane >> 4;
  const int r0   = wid * 16;
  constexpr int KK = Hh * Dd;  // 512

  const _Float16* Hr = heads + (size_t)r0 * KK;
  const _Float16* Wh = Wsw + (size_t)hf * 1024;

  v8f acc[4] = {};
  v16h a = load_a_f16(Hr, KK);
  v16h b[4];
#pragma unroll
  for (int t = 0; t < 4; ++t)
    b[t] = *(const v16h*)(Wh + (t * 16 + col) * 16);

#pragma unroll
  for (int k0 = 0; k0 < KK; k0 += 32) {
    const int kn = (k0 + 32 < KK) ? (k0 + 32) : k0;
    v16h an = load_a_f16(Hr + kn, KK);
    const _Float16* wn = Wh + (size_t)(kn >> 5) * 2048;
    v16h bn[4];
#pragma unroll
    for (int t = 0; t < 4; ++t)
      bn[t] = *(const v16h*)(wn + (t * 16 + col) * 16);
#pragma unroll
    for (int t = 0; t < 4; ++t) acc[t] = WMMA_F16(a, b[t], acc[t]);
    a = an;
#pragma unroll
    for (int t = 0; t < 4; ++t) b[t] = bn[t];
  }

#pragma unroll
  for (int t = 0; t < 4; ++t)
#pragma unroll
    for (int j = 0; j < 8; ++j)
      outp[(size_t)(r0 + j + hf * 8) * Dd + t * 16 + col] = acc[t][j];
}

// ---------------------------------------------------------------------------
extern "C" void kernel_launch(void* const* d_in, const int* in_sizes, int n_in,
                              void* d_out, int out_size, void* d_ws, size_t ws_size,
                              hipStream_t stream) {
  (void)in_sizes; (void)n_in; (void)out_size; (void)ws_size;
  const float* query = (const float*)d_in[0];
  const float* key   = (const float*)d_in[1];
  const float* value = (const float*)d_in[2];
  const float* WQ    = (const float*)d_in[3];
  const float* WK    = (const float*)d_in[4];
  const float* WV    = (const float*)d_in[5];
  const float* WO    = (const float*)d_in[6];

  const size_t nelem = (size_t)Bb * Hh * Ss * Dd;  // 4M f16 each
  const size_t xelem = (size_t)Bb * Ss * QF;       // 4M f16 each
  const size_t welem = (size_t)Hh * QF * Dd;       // 256K f16 each
  _Float16* p = (_Float16*)d_ws;
  _Float16* qws = p; p += nelem;
  _Float16* kws = p; p += nelem;
  _Float16* vws = p; p += nelem;   // transposed [b,h,d,s]
  _Float16* hws = p; p += nelem;   // heads [b,s,h*64+v]
  _Float16* xq  = p; p += xelem;
  _Float16* xk  = p; p += xelem;
  _Float16* xv  = p; p += xelem;
  _Float16* wqs = p; p += welem;
  _Float16* wks = p; p += welem;
  _Float16* wvs = p; p += welem;
  _Float16* wos = p; p += (size_t)QF * Dd;

  // --- one-time converts / swizzles (all cheap, bandwidth-bound) -----------
  cvt_f16_kernel<<<(int)(xelem / (256 * 4)), 256, 0, stream>>>(query, xq);
  cvt_f16_kernel<<<(int)(xelem / (256 * 4)), 256, 0, stream>>>(key,   xk);
  cvt_f16_kernel<<<(int)(xelem / (256 * 4)), 256, 0, stream>>>(value, xv);
  wswz_kernel<<<(int)(welem / 256), 256, 0, stream>>>(WQ, wqs);
  wswz_kernel<<<(int)(welem / 256), 256, 0, stream>>>(WK, wks);
  wswz_kernel<<<(int)(welem / 256), 256, 0, stream>>>(WV, wvs);
  wswz_kernel<<<(int)((size_t)QF * Dd / 256), 256, 0, stream>>>(WO, wos);

  // --- projections: 4096 wave-tiles each, 8 waves/block ---------------------
  proj_kernel<<<512, 256, 0, stream>>>(xq, wqs, qws, 0.125f, 0); // q pre-scaled
  proj_kernel<<<512, 256, 0, stream>>>(xk, wks, kws, 1.0f,   0);
  proj_kernel<<<512, 256, 0, stream>>>(xv, wvs, vws, 1.0f,   1);

  // --- attention: B*H*(S/128) = 512 blocks ----------------------------------
  attn_kernel<<<512, 256, 0, stream>>>(qws, kws, vws, hws);

  // --- output projection: 512 row tiles / 8 waves per block -----------------
  oproj_kernel<<<64, 256, 0, stream>>>(hws, wos, (float*)d_out);
}
